// CtrlPointHungarianMatcher_23158463660802
// MI455X (gfx1250) — matrix operand order repaired and checked by reading, not compiled, not run on hardware
//
#include <hip/hip_runtime.h>

#define BS 16
#define NQ 200
#define NQP 208          // 13 * 16, padded M for WMMA tiling
#define MT 13            // M tiles per batch
#define NPTS 25
#define NGT 32
#define LTXT 25
#define VOC 96
#define VOCP1 97
#define ALPHA_F 0.25f
#define THR_F 0.85f
#define KL_EPS 1e-6f
#define FOCAL_EPS 1e-8f

typedef __attribute__((ext_vector_type(16))) _Float16 v16h;
typedef __attribute__((ext_vector_type(8)))  _Float16 v8h;
typedef __attribute__((ext_vector_type(8)))  float    v8f;

// ---------------- block reductions (blockDim.x == NT) ----------------
template<int NT>
__device__ inline float blkmax(float v, float* sm) {
    int t = threadIdx.x;
    sm[t] = v; __syncthreads();
    #pragma unroll
    for (int s = NT / 2; s > 0; s >>= 1) {
        if (t < s) sm[t] = fmaxf(sm[t], sm[t + s]);
        __syncthreads();
    }
    float r = sm[0]; __syncthreads();
    return r;
}

template<int NT>
__device__ inline float blksum(float v, float* sm) {
    int t = threadIdx.x;
    sm[t] = v; __syncthreads();
    #pragma unroll
    for (int s = NT / 2; s > 0; s >>= 1) {
        if (t < s) sm[t] += sm[t + s];
        __syncthreads();
    }
    float r = sm[0]; __syncthreads();
    return r;
}

// ---------------- A: focal classification cost per query ----------------
__global__ __launch_bounds__(256) void focal_kernel(const float* __restrict__ pl,
                                                    float* __restrict__ cc) {
    int idx = blockIdx.x * blockDim.x + threadIdx.x;
    if (idx >= BS * NQ) return;
    const float* p = pl + (size_t)idx * NPTS;
    float s = 0.f;
    #pragma unroll
    for (int i = 0; i < NPTS; ++i) {
        float x  = p[i];
        float pr = 1.f / (1.f + __expf(-x));
        float pos = ALPHA_F * (1.f - pr) * (1.f - pr) * (-__logf(pr + FOCAL_EPS));
        float neg = (1.f - ALPHA_F) * pr * pr * (-__logf(1.f - pr + FOCAL_EPS));
        s += pos - neg;
    }
    cc[idx] = s * (1.f / (float)NPTS);
}

// ---------------- B0: centroid inverse norms ----------------
__global__ __launch_bounds__(256) void norm_kernel(const float* __restrict__ cen,
                                                   float* __restrict__ invn) {
    __shared__ float sm[256];
    int i = blockIdx.x, t = threadIdx.x;
    float v = cen[(size_t)i * 256 + t];
    float ss = blksum<256>(v * v, sm);
    if (t == 0) invn[i] = rsqrtf(fmaxf(ss, 1e-30f));
}

// ---------------- B1: soft_all = THR*I + (1-THR)*softmax(cn@cn^T) ----------------
__global__ __launch_bounds__(128) void softall_kernel(const float* __restrict__ cen,
                                                      const float* __restrict__ invn,
                                                      float* __restrict__ soft) {
    __shared__ float row[256];
    __shared__ float sm[128];
    int i = blockIdx.x, t = threadIdx.x;
    row[t]       = cen[(size_t)i * 256 + t];
    row[t + 128] = cen[(size_t)i * 256 + t + 128];
    __syncthreads();
    float s = -1e30f;
    if (t < VOC) {
        const float* cj = cen + (size_t)t * 256;
        float d = 0.f;
        #pragma unroll 8
        for (int k = 0; k < 256; ++k) d += row[k] * cj[k];
        s = d * invn[i] * invn[t];
    }
    float m   = blkmax<128>(s, sm);
    float e   = (t < VOC) ? __expf(s - m) : 0.f;
    float sum = blksum<128>(e, sm);
    if (t < VOC)
        soft[i * VOC + t] = THR_F * ((i == t) ? 1.f : 0.f) + (1.f - THR_F) * (e / sum);
}

// ---------------- C: per-(b,g) soft target distribution T, T*logT, lengths ----------------
__global__ __launch_bounds__(128) void tgt_kernel(const int* __restrict__ txt,
                                                  const float* __restrict__ soft,
                                                  _Float16* __restrict__ T16,
                                                  float* __restrict__ tlogt,
                                                  int* __restrict__ lens) {
    __shared__ float sm[128];
    __shared__ int chars[LTXT];
    int bg = blockIdx.x, t = threadIdx.x;
    if (t < LTXT) chars[t] = txt[(size_t)bg * LTXT + t];
    __syncthreads();
    int len = 0;
    float acc = 0.f;
    for (int l = 0; l < LTXT; ++l) {
        int c = chars[l];
        if (c != VOC) {
            ++len;
            if (t < VOC) acc += soft[min(c, VOC - 1) * VOC + t];
        }
    }
    float avg = acc / (float)max(len, 1);
    float Tv = (t < VOC) ? (avg + KL_EPS) : 0.f;
    float denom = blksum<128>(Tv, sm);
    float Tn = Tv / denom;
    float tl = (t < VOC) ? Tn * __logf(Tn) : 0.f;
    float tls = blksum<128>(tl, sm);
    if (t < VOC) T16[(size_t)bg * VOC + t] = (_Float16)Tn;
    if (t == 0) { tlogt[bg] = tls; lens[bg] = len; }
}

// ---------------- D: logP from pred_text_logits (softmax + mean + renorm) ----------------
__global__ __launch_bounds__(128) void logp_kernel(const float* __restrict__ ptl,
                                                   _Float16* __restrict__ lp16) {
    __shared__ float sm[128];
    int bqp = blockIdx.x;
    int b = bqp / NQP, q = bqp % NQP;
    int t = threadIdx.x;
    _Float16* out = lp16 + ((size_t)b * NQP + q) * VOC;
    if (q >= NQ) {                     // zero the M padding rows
        if (t < VOC) out[t] = (_Float16)0.f;
        return;
    }
    const float* base = ptl + (size_t)(b * NQ + q) * NPTS * VOCP1;
    float avg = 0.f;
    for (int p = 0; p < NPTS; ++p) {
        float x = (t < VOCP1) ? base[p * VOCP1 + t] : -1e30f;
        float m = blkmax<128>(x, sm);
        float e = (t < VOCP1) ? __expf(x - m) : 0.f;
        float s = blksum<128>(e, sm);
        avg += e / s;
    }
    avg *= (1.f / (float)NPTS);
    float p96 = (t < VOC) ? avg : 0.f;
    float s96 = blksum<128>(p96, sm);
    float denom = s96 + (float)VOC * KL_EPS;
    if (t < VOC) out[t] = (_Float16)(__logf((avg + KL_EPS) / denom));
}

// ---------------- E: WMMA kernel — kl[b,q,g] = T*logT - logP . T ----------------
// One 8-wave workgroup per batch. The 32x96 f16 T-tile (reused by all 13
// M-tiles) is staged into LDS with the CDNA5 async copy engine
// (global_load_async_to_lds_b128 + s_wait_asynccnt), then each wave computes
// 16x32 output tiles with v_wmma_f32_16x16x32_f16 (B fragments from LDS,
// A fragments streamed from global — A has no reuse).
__global__ __launch_bounds__(256) void kl_wmma_kernel(const _Float16* __restrict__ lp16,
                                                      const _Float16* __restrict__ T16,
                                                      const float* __restrict__ tlogt,
                                                      const int* __restrict__ lens,
                                                      float* __restrict__ klbuf) {
    alignas(32) __shared__ _Float16 Bsh[NGT * VOC];   // 6 KB
    const int b   = blockIdx.x;
    const int tid = threadIdx.x;

    // ---- async stage: 384 x b128 chunks, wave-uniform guard ----
    {
        const _Float16* src = T16 + (size_t)b * NGT * VOC;
        #pragma unroll
        for (int r = 0; r < 2; ++r) {
            int idx = tid + r * 256;                       // need 0..383
            if (idx < (NGT * VOC / 8)) {
                unsigned lds_off = (unsigned)(size_t)(&Bsh[idx * 8]);
                unsigned long long gaddr =
                    (unsigned long long)(size_t)(src + idx * 8);
                asm volatile("global_load_async_to_lds_b128 %0, %1, off"
                             :: "v"(lds_off), "v"(gaddr) : "memory");
            }
        }
        asm volatile("s_wait_asynccnt 0" ::: "memory");
    }
    __syncthreads();

    const int wave = tid >> 5;
    const int lane = tid & 31;
    const int n  = lane & 15;      // A-row index for loads / B,D column index
    const int hi = lane >> 4;

    for (int mt = wave; mt < MT; mt += 8) {            // wave-uniform tiles
        const _Float16* Arow = lp16 + ((size_t)b * NQP + mt * 16 + n) * VOC;
        const _Float16* B0 = &Bsh[(size_t)n * VOC];
        const _Float16* B1 = &Bsh[(size_t)(16 + n) * VOC];

        v8f acc0 = {}, acc1 = {};
        #pragma unroll
        for (int ks = 0; ks < 3; ++ks) {
            int k0 = ks * 32;
            // A fragment: 16 contiguous k starting at k0 + hi*16
            v16h a = *(const v16h*)(Arow + k0 + hi * 16);
            // B fragment from LDS: 16 contiguous k starting at k0 + hi*16
            v16h b0 = *(const v16h*)(B0 + k0 + hi * 16);
            v16h b1 = *(const v16h*)(B1 + k0 + hi * 16);
            acc0 = __builtin_amdgcn_wmma_f32_16x16x32_f16(false, a, false, b0,
                                                          (short)0, acc0, false, false);
            acc1 = __builtin_amdgcn_wmma_f32_16x16x32_f16(false, a, false, b1,
                                                          (short)0, acc1, false, false);
        }

        // D layout: element r of v8f holds row m = hi*8 + r, column n
        #pragma unroll
        for (int r = 0; r < 8; ++r) {
            int q = mt * 16 + hi * 8 + r;
            if (q < NQ) {
                int g0 = n, g1 = 16 + n;
                float kl0 = tlogt[b * NGT + g0] - acc0[r];
                float kl1 = tlogt[b * NGT + g1] - acc1[r];
                if (lens[b * NGT + g0] == 0) kl0 = 100.f;
                if (lens[b * NGT + g1] == 0) kl1 = 100.f;
                size_t o = (size_t)(b * NQ + q) * NGT;
                klbuf[o + g0] = kl0;
                klbuf[o + g1] = kl1;
            }
        }
    }
}

// ---------------- F: final assembly: cc + L1 cdist + block-diag KL ----------------
__global__ __launch_bounds__(256) void final_kernel(const float* __restrict__ predp,
                                                    const float* __restrict__ tgtp,
                                                    const float* __restrict__ cc,
                                                    const float* __restrict__ klbuf,
                                                    float* __restrict__ out) {
    __shared__ float o[2 * NPTS];
    int bq = blockIdx.x;
    int b = bq / NQ;
    int t = threadIdx.x;
    if (t < 2 * NPTS) o[t] = predp[(size_t)bq * (2 * NPTS) + t];
    __syncthreads();
    float c0 = cc[bq];
    for (int col = t; col < BS * NGT; col += 256) {
        const float* tp = tgtp + (size_t)col * (2 * NPTS);
        float l1 = 0.f;
        #pragma unroll
        for (int k = 0; k < 2 * NPTS; ++k) l1 += fabsf(o[k] - tp[k]);
        float val = c0 + l1;
        if ((col >> 5) == b) val += klbuf[(size_t)bq * NGT + (col & 31)];
        out[(size_t)bq * (BS * NGT) + col] = val;
    }
}

// ---------------- launch ----------------
extern "C" void kernel_launch(void* const* d_in, const int* in_sizes, int n_in,
                              void* d_out, int out_size, void* d_ws, size_t ws_size,
                              hipStream_t stream) {
    (void)in_sizes; (void)n_in; (void)out_size; (void)ws_size;
    const float* pred_logits = (const float*)d_in[0];
    const float* pred_ctrl   = (const float*)d_in[1];
    const float* pred_text   = (const float*)d_in[2];
    const float* tgt_ctrl    = (const float*)d_in[3];
    const int*   tgt_texts   = (const int*)d_in[4];
    const float* centroids   = (const float*)d_in[5];
    float* out = (float*)d_out;

    // workspace carve-out (256B aligned segments; ~1.2 MB total)
    char* ws = (char*)d_ws;
    size_t off = 0;
    auto carve = [&](size_t bytes) -> char* {
        char* p = ws + off;
        off = (off + bytes + 255) & ~(size_t)255;
        return p;
    };
    float*    invn  = (float*)carve((size_t)VOC * 4);
    float*    soft  = (float*)carve((size_t)VOC * VOC * 4);
    float*    tlogt = (float*)carve((size_t)BS * NGT * 4);
    int*      lens  = (int*)carve((size_t)BS * NGT * 4);
    float*    cc    = (float*)carve((size_t)BS * NQ * 4);
    float*    klbuf = (float*)carve((size_t)BS * NQ * NGT * 4);
    _Float16* T16   = (_Float16*)carve((size_t)BS * NGT * VOC * 2);
    _Float16* lp16  = (_Float16*)carve((size_t)BS * NQP * VOC * 2);

    focal_kernel<<<(BS * NQ + 255) / 256, 256, 0, stream>>>(pred_logits, cc);
    norm_kernel<<<VOC, 256, 0, stream>>>(centroids, invn);
    softall_kernel<<<VOC, 128, 0, stream>>>(centroids, invn, soft);
    tgt_kernel<<<BS * NGT, 128, 0, stream>>>(tgt_texts, soft, T16, tlogt, lens);
    logp_kernel<<<BS * NQP, 128, 0, stream>>>(pred_text, lp16);
    kl_wmma_kernel<<<BS, 256, 0, stream>>>(lp16, T16, tlogt, lens, klbuf);
    final_kernel<<<BS * NQ, 256, 0, stream>>>(pred_ctrl, tgt_ctrl, cc, klbuf, out);
}
